// Exaone4Attention_5506148073494
// MI455X (gfx1250) — matrix operand order repaired and checked by reading, not compiled, not run on hardware
//
#include <hip/hip_runtime.h>
#include <hip/hip_bf16.h>

// ---------------------------------------------------------------------------
// Exaone4 sliding-window attention for MI455X (gfx1250, wave32, WMMA bf16)
// ---------------------------------------------------------------------------
// Pipeline:
//   1) fp32 -> bf16 convert: hidden, wqkv, wo            (bandwidth, one-shot)
//   2) qkv = hidden @ wqkv^T        (WMMA bf16, fp32 acc, double-buffered K)
//   3) RMSNorm(q,k) + NeoX RoPE(q,k); V transposed to [HK][D][T] bf16
//   4) flash attention, 1 wave per (head, 16-row q tile), S-step 32,
//      scores + P*V both on v_wmma_f32_16x16x32_bf16
//   5) out = attn @ wo^T            (WMMA bf16, fp32 acc -> d_out fp32)
// ---------------------------------------------------------------------------

typedef __bf16 bf16;
typedef __attribute__((ext_vector_type(16))) __bf16 v16bf;
typedef __attribute__((ext_vector_type(8)))  __bf16 v8bf;
typedef __attribute__((ext_vector_type(8)))  float  v8f;

#define T_     2048
#define HID_   4096
#define H_     32
#define HK_    8
#define D_     128
#define WIN_   1023
#define NQKV_  6144          // (H + 2*HK) * D
#define SCALE_ 0.08838834764831845f   // D^-0.5
#define NEGBIG_ (-1.0e30f)

// ---------------------------------------------------------------------------
// fp32 -> bf16 elementwise convert
// ---------------------------------------------------------------------------
__global__ __launch_bounds__(256) void cvt_f32_bf16(const float* __restrict__ in,
                                                    bf16* __restrict__ out, size_t n) {
    size_t i = (size_t)blockIdx.x * 256 + threadIdx.x;
    if (i < n) out[i] = (bf16)in[i];
}

// ---------------------------------------------------------------------------
// WMMA fragment loaders (16x16x32 bf16; this code's contiguous K order, same for A and B)
// A (16x32, MxK): lane holds row (lane&15); 16 contiguous K at hi*16
// B (32x16, KxN): lane holds col (lane&15); 16 contiguous K at hi*16
//   (B supplied as B^T row-major [N][K], so each lane reads 32 contiguous bytes)
// ---------------------------------------------------------------------------
__device__ inline v16bf load_a_frag(const bf16* __restrict__ A, int lda,
                                    int row_base, int k_base, int lane) {
    int r  = lane & 15;
    int hi = lane >> 4;
    const bf16* p = A + (size_t)(row_base + r) * lda + k_base + hi * 16;
    v8bf lo = *(const v8bf*)p;
    v8bf hv = *(const v8bf*)(p + 8);
    v16bf a;
#pragma unroll
    for (int i = 0; i < 8; ++i) { a[i] = lo[i]; a[8 + i] = hv[i]; }
    return a;
}

__device__ inline v16bf load_b_frag(const bf16* __restrict__ Bt, int ldb,
                                    int col_base, int k_base, int lane) {
    int n  = lane & 15;
    int hi = lane >> 4;
    return *(const v16bf*)(Bt + (size_t)(col_base + n) * ldb + k_base + hi * 16);
}

// ---------------------------------------------------------------------------
// GEMM: C[M,N] (fp32) = A[M,K] (bf16) @ Bt[N,K]^T (bf16)
// Block = 256 threads = 8 waves, 128x128 tile; wave grid 4(M) x 2(N),
// each wave 32x64 => 2x4 C fragments; K step 32, software-pipelined x2 so
// the 8 loads of stage s+1 are in flight while the 8 WMMAs of stage s issue.
// ---------------------------------------------------------------------------
__global__ __launch_bounds__(256) void gemm_bf16_wmma(const bf16* __restrict__ A,
                                                      const bf16* __restrict__ Bt,
                                                      float* __restrict__ C,
                                                      int M, int N, int K) {
    int lane = threadIdx.x & 31;
    int wave = threadIdx.x >> 5;
    int wm = wave & 3;                 // 0..3
    int wn = wave >> 2;                // 0..1
    int m0 = blockIdx.x * 128 + wm * 32;
    int n0 = blockIdx.y * 128 + wn * 64;

    v8f acc[2][4] = {};

    v16bf a0[2], b0[4], a1[2], b1[4];

    // prologue: stage 0 loads
#pragma unroll
    for (int i = 0; i < 2; ++i) a0[i] = load_a_frag(A, K, m0 + i * 16, 0, lane);
#pragma unroll
    for (int j = 0; j < 4; ++j) b0[j] = load_b_frag(Bt, K, n0 + j * 16, 0, lane);

    for (int kk = 0; kk < K; kk += 64) {
        // issue stage-1 loads (kk+32) before consuming stage 0
#pragma unroll
        for (int i = 0; i < 2; ++i) a1[i] = load_a_frag(A, K, m0 + i * 16, kk + 32, lane);
#pragma unroll
        for (int j = 0; j < 4; ++j) b1[j] = load_b_frag(Bt, K, n0 + j * 16, kk + 32, lane);
        __builtin_prefetch(Bt + (size_t)(n0 + (lane & 15)) * K + kk + 128, 0, 1);

#pragma unroll
        for (int i = 0; i < 2; ++i)
#pragma unroll
            for (int j = 0; j < 4; ++j)
                acc[i][j] = __builtin_amdgcn_wmma_f32_16x16x32_bf16(
                    false, a0[i], false, b0[j], (short)0, acc[i][j], false, false);

        // issue stage-0 loads for next iteration (kk+64) before consuming stage 1
        if (kk + 64 < K) {
#pragma unroll
            for (int i = 0; i < 2; ++i) a0[i] = load_a_frag(A, K, m0 + i * 16, kk + 64, lane);
#pragma unroll
            for (int j = 0; j < 4; ++j) b0[j] = load_b_frag(Bt, K, n0 + j * 16, kk + 64, lane);
            __builtin_prefetch(A + (size_t)(m0 + (lane & 15)) * K + kk + 160, 0, 1);
        }

#pragma unroll
        for (int i = 0; i < 2; ++i)
#pragma unroll
            for (int j = 0; j < 4; ++j)
                acc[i][j] = __builtin_amdgcn_wmma_f32_16x16x32_bf16(
                    false, a1[i], false, b1[j], (short)0, acc[i][j], false, false);
    }

    int n  = lane & 15;
    int hi = lane >> 4;
#pragma unroll
    for (int i = 0; i < 2; ++i)
#pragma unroll
        for (int j = 0; j < 4; ++j)
#pragma unroll
            for (int r = 0; r < 8; ++r) {
                int row = m0 + i * 16 + r + 8 * hi;
                int col = n0 + j * 16 + n;
                C[(size_t)row * N + col] = acc[i][j][r];
            }
}

// ---------------------------------------------------------------------------
// RMSNorm + RoPE (NeoX) on q/k heads; v converted + transposed.
// One block (128 thr) per (head_slot, token). head_slot: 0..31 q, 32..39 k,
// 40..47 v.  Outputs: q_bf16[T][H][D], k_bf16[HK][T][D], vT_bf16[HK][D][T].
// ---------------------------------------------------------------------------
__global__ __launch_bounds__(128) void norm_rope_kernel(const float* __restrict__ qkv,
                                                        const int*   __restrict__ pos,
                                                        const float* __restrict__ qw,
                                                        const float* __restrict__ kw,
                                                        bf16* __restrict__ qb,
                                                        bf16* __restrict__ kb,
                                                        bf16* __restrict__ vtb) {
    int hh = blockIdx.x;     // 0..47
    int t  = blockIdx.y;     // 0..T-1
    int d  = threadIdx.x;    // 0..127

    const float* x = qkv + (size_t)t * NQKV_ + (size_t)hh * D_;
    float v = x[d];

    if (hh >= 40) {          // V head: just convert + transpose
        int kvh = hh - 40;
        vtb[((size_t)kvh * D_ + d) * T_ + t] = (bf16)v;
        return;
    }

    __shared__ float red[128];
    red[d] = v * v;
    __syncthreads();
#pragma unroll
    for (int s = 64; s > 0; s >>= 1) {
        if (d < s) red[d] += red[d + s];
        __syncthreads();
    }
    float rms = rsqrtf(red[0] / (float)D_ + 1e-6f);
    __syncthreads();

    float wgt = (hh < 32) ? qw[d] : kw[d];
    float xn  = v * rms * wgt;
    red[d] = xn;
    __syncthreads();

    // NeoX RoPE
    int   i  = d & 63;
    float fr = (float)pos[t] * __powf(1.0e6f, -((float)i) / 64.0f);
    float c  = cosf(fr);
    float s2 = sinf(fr);
    float x1 = red[i];
    float x2 = red[i + 64];
    float out = (d < 64) ? (x1 * c - x2 * s2) : (x2 * c + x1 * s2);

    if (hh < 32) {
        qb[((size_t)t * H_ + hh) * D_ + d] = (bf16)out;
    } else {
        int kvh = hh - 32;
        kb[((size_t)kvh * T_ + t) * D_ + d] = (bf16)out;
    }
}

// ---------------------------------------------------------------------------
// Flash attention. 1 wave per (head, 16-row q tile); 4 waves per block.
// S-step = 32 keys: scores via 8 wmma (2 tiles x K=128/32), online softmax
// through per-wave LDS (C layout -> rows -> A layout), P*V via 8 wmma.
// Output written bf16 to attn[T][H*D] for the final projection GEMM.
// ---------------------------------------------------------------------------
#define AW 4   // waves per block
__global__ __launch_bounds__(128) void attn_kernel(const bf16* __restrict__ qb,
                                                   const bf16* __restrict__ kb,
                                                   const bf16* __restrict__ vtb,
                                                   bf16* __restrict__ ob) {
    __shared__ float s_sc[AW][16][32];
    __shared__ bf16  s_p[AW][16][32];
    __shared__ float s_alpha[AW][16];
    __shared__ float s_inv[AW][16];

    int lane = threadIdx.x & 31;
    int w    = threadIdx.x >> 5;
    int wid  = blockIdx.x * AW + w;
    int h    = wid >> 7;            // / (T/16)
    int qt   = wid & 127;
    int kvh  = h >> 2;              // GQA group of 4
    int r15  = lane & 15;
    int hi   = lane >> 4;
    int qlo  = qt * 16;

    const bf16* kbase = kb  + (size_t)kvh * T_ * D_;
    const bf16* vbase = vtb + (size_t)kvh * D_ * T_;

    // Q fragments for the whole 16x128 tile (A layout), 4 K-chunks of 32
    v16bf qa[4];
#pragma unroll
    for (int kbk = 0; kbk < 4; ++kbk) {
        const bf16* p = qb + ((size_t)(qlo + r15) * H_ + h) * D_ + kbk * 32 + hi * 16;
        v8bf lo = *(const v8bf*)p;
        v8bf hv = *(const v8bf*)(p + 8);
#pragma unroll
        for (int i = 0; i < 8; ++i) { qa[kbk][i] = lo[i]; qa[kbk][8 + i] = hv[i]; }
    }

    v8f   o[8] = {};
    float m_i = NEGBIG_, l_i = 0.f;

    int s_begin = qlo - WIN_;
    if (s_begin < 0) s_begin = 0;
    s_begin &= ~31;
    int s_end = qlo + 16;           // last key index used is qlo+15

    for (int sb = s_begin; sb < s_end; sb += 32) {
        // ---- scores: two 16x16 tiles over K=128 ----
        v8f c0 = {}, c1 = {};
#pragma unroll
        for (int kbk = 0; kbk < 4; ++kbk) {
            v16bf b0 = *(const v16bf*)(kbase + (size_t)(sb + r15) * D_      + kbk * 32 + hi * 16);
            v16bf b1 = *(const v16bf*)(kbase + (size_t)(sb + 16 + r15) * D_ + kbk * 32 + hi * 16);
            c0 = __builtin_amdgcn_wmma_f32_16x16x32_bf16(false, qa[kbk], false, b0,
                                                         (short)0, c0, false, false);
            c1 = __builtin_amdgcn_wmma_f32_16x16x32_bf16(false, qa[kbk], false, b1,
                                                         (short)0, c1, false, false);
        }

        // ---- scale + sliding-window causal mask -> LDS (C layout) ----
#pragma unroll
        for (int r = 0; r < 8; ++r) {
            int Mr = r + 8 * hi;
            int qi = qlo + Mr;
            int kj0 = sb + r15;
            int kj1 = kj0 + 16;
            float v0 = (kj0 <= qi && (qi - kj0) <= WIN_) ? c0[r] * SCALE_ : NEGBIG_;
            float v1 = (kj1 <= qi && (qi - kj1) <= WIN_) ? c1[r] * SCALE_ : NEGBIG_;
            s_sc[w][Mr][r15]      = v0;
            s_sc[w][Mr][16 + r15] = v1;
        }

        // ---- online softmax: lanes 0..15 each own one row ----
        if (lane < 16) {
            float bmax = NEGBIG_;
#pragma unroll
            for (int j = 0; j < 32; ++j) bmax = fmaxf(bmax, s_sc[w][lane][j]);
            float mnew  = fmaxf(m_i, bmax);
            float alpha = __expf(m_i - mnew);
            float sum = 0.f;
#pragma unroll
            for (int j = 0; j < 32; ++j) {
                float sc = s_sc[w][lane][j];
                float p  = (sc > -1.0e29f) ? __expf(sc - mnew) : 0.f;
                sum += p;
                s_p[w][lane][j] = (bf16)p;
            }
            l_i = l_i * alpha + sum;
            m_i = mnew;
            s_alpha[w][lane] = alpha;
        }

        // ---- rescale running output ----
        float al[8];
#pragma unroll
        for (int r = 0; r < 8; ++r) al[r] = s_alpha[w][r + 8 * hi];
#pragma unroll
        for (int f = 0; f < 8; ++f)
#pragma unroll
            for (int r = 0; r < 8; ++r) o[f][r] *= al[r];

        // ---- P (A layout, K=32) from LDS ----
        v16bf pa;
        {
            const bf16* pp = &s_p[w][r15][hi * 16];
            v8bf lo = *(const v8bf*)pp;
            v8bf hv = *(const v8bf*)(pp + 8);
#pragma unroll
            for (int i = 0; i < 8; ++i) { pa[i] = lo[i]; pa[8 + i] = hv[i]; }
        }

        // ---- O += P @ V  (V^T gives contiguous B fragments) ----
#pragma unroll
        for (int dt = 0; dt < 8; ++dt) {
            const bf16* vp = vbase + (size_t)(dt * 16 + r15) * T_ + sb + hi * 16;
            v16bf bv = *(const v16bf*)vp;
            o[dt] = __builtin_amdgcn_wmma_f32_16x16x32_bf16(false, pa, false, bv,
                                                            (short)0, o[dt], false, false);
        }
    }

    // ---- normalize + store bf16 ----
    if (lane < 16) s_inv[w][lane] = (l_i > 0.f) ? (1.0f / l_i) : 0.f;
    float inv[8];
#pragma unroll
    for (int r = 0; r < 8; ++r) inv[r] = s_inv[w][r + 8 * hi];
#pragma unroll
    for (int dt = 0; dt < 8; ++dt)
#pragma unroll
        for (int r = 0; r < 8; ++r) {
            int t = qlo + r + 8 * hi;
            ob[(size_t)t * (H_ * D_) + h * D_ + dt * 16 + r15] = (bf16)(o[dt][r] * inv[r]);
        }
}

// ---------------------------------------------------------------------------
// Host-side launcher
// ---------------------------------------------------------------------------
static inline size_t align256(size_t x) { return (x + 255) & ~(size_t)255; }

extern "C" void kernel_launch(void* const* d_in, const int* in_sizes, int n_in,
                              void* d_out, int out_size, void* d_ws, size_t ws_size,
                              hipStream_t stream) {
    (void)in_sizes; (void)n_in; (void)out_size; (void)ws_size;

    const int*   positions = (const int*)d_in[0];
    const float* hidden    = (const float*)d_in[1];
    const float* wqkv      = (const float*)d_in[2];
    const float* wo        = (const float*)d_in[3];
    const float* q_norm_w  = (const float*)d_in[4];
    const float* k_norm_w  = (const float*)d_in[5];
    float*       out       = (float*)d_out;

    // Workspace carve-out
    char* ws = (char*)d_ws;
    size_t off = 0;
    bf16*  hb    = (bf16*)(ws + off); off += align256((size_t)T_ * HID_ * sizeof(bf16));
    bf16*  wqkvb = (bf16*)(ws + off); off += align256((size_t)NQKV_ * HID_ * sizeof(bf16));
    bf16*  wob   = (bf16*)(ws + off); off += align256((size_t)HID_ * (H_ * D_) * sizeof(bf16));
    float* qkvf  = (float*)(ws + off); off += align256((size_t)T_ * NQKV_ * sizeof(float));
    bf16*  qb    = (bf16*)(ws + off); off += align256((size_t)T_ * H_ * D_ * sizeof(bf16));
    bf16*  kbuf  = (bf16*)(ws + off); off += align256((size_t)HK_ * T_ * D_ * sizeof(bf16));
    bf16*  vtb   = (bf16*)(ws + off); off += align256((size_t)HK_ * D_ * T_ * sizeof(bf16));
    bf16*  attnb = (bf16*)(ws + off); off += align256((size_t)T_ * H_ * D_ * sizeof(bf16));

    // 1) fp32 -> bf16 conversions
    {
        size_t n;
        n = (size_t)T_ * HID_;
        cvt_f32_bf16<<<(unsigned)((n + 255) / 256), 256, 0, stream>>>(hidden, hb, n);
        n = (size_t)NQKV_ * HID_;
        cvt_f32_bf16<<<(unsigned)((n + 255) / 256), 256, 0, stream>>>(wqkv, wqkvb, n);
        n = (size_t)HID_ * (H_ * D_);
        cvt_f32_bf16<<<(unsigned)((n + 255) / 256), 256, 0, stream>>>(wo, wob, n);
    }

    // 2) qkv = hidden @ wqkv^T   (M=2048, N=6144, K=4096)
    {
        dim3 grid(T_ / 128, NQKV_ / 128);
        gemm_bf16_wmma<<<grid, 256, 0, stream>>>(hb, wqkvb, qkvf, T_, NQKV_, HID_);
    }

    // 3) RMSNorm + RoPE; V transpose
    {
        dim3 grid(H_ + 2 * HK_, T_);   // 48 x 2048
        norm_rope_kernel<<<grid, 128, 0, stream>>>(qkvf, positions, q_norm_w, k_norm_w,
                                                   qb, kbuf, vtb);
    }

    // 4) flash attention -> attnb (bf16 [T][H*D])
    {
        int n_waves = H_ * (T_ / 16);          // 4096
        attn_kernel<<<n_waves / AW, AW * 32, 0, stream>>>(qb, kbuf, vtb, attnb);
    }

    // 5) out = attnb @ wo^T   (M=2048, N=4096, K=4096) -> fp32 d_out
    {
        dim3 grid(T_ / 128, (H_ * D_) / 128);
        gemm_bf16_wmma<<<grid, 256, 0, stream>>>(attnb, wob, out, T_, H_ * D_, HID_);
    }
}
